// gaoModel_19250043420939
// MI455X (gfx1250) — matrix operand order repaired and checked by reading, not compiled, not run on hardware
//
#include <hip/hip_runtime.h>

typedef __bf16 bf16_t;
typedef bf16_t v16bf __attribute__((ext_vector_type(16)));
typedef float  v8f   __attribute__((ext_vector_type(8)));
typedef float  v2f   __attribute__((ext_vector_type(2)));

#define B_   16
#define H_   512
#define T0_  400
#define F0_  40
#define T1_  396
#define T2_  392
#define T3_  384
#define T1P_ 464   // deep zero-pad so conv2's 4-tile groups stay in bounds
#define T2P_ 400
#define KSQ_ (H_ * H_)   // 262144

// ---------- helpers ----------

__device__ __forceinline__ bf16_t f2bf(float x) {
  union { float f; unsigned u; } v; v.f = x;
  unsigned u = v.u;
  u += 0x7FFFu + ((u >> 16) & 1u);           // round-to-nearest-even
  union { unsigned short s; bf16_t b; } o;
  o.s = (unsigned short)(u >> 16);
  return o.b;
}

// A fragment: row-major A[M][K], 16x32 bf16 tile at (row0, k0).
__device__ __forceinline__ v16bf load_a_frag(const bf16_t* __restrict__ A, int lda,
                                             int row0, int k0) {
  int lane = threadIdx.x & 31;
  int hi = lane >> 4;
  const bf16_t* p = A + (size_t)(row0 + (lane & 15)) * lda + k0 + hi * 16;
  v16bf a;
#pragma unroll
  for (int j = 0; j < 8; ++j) { a[j] = p[j]; a[8 + j] = p[8 + j]; }
  return a;
}

// B fragment: B stored column-major (Bc[col][k] contiguous over k), 32x16 tile.
__device__ __forceinline__ v16bf load_b_frag(const bf16_t* __restrict__ Bc, int ldb,
                                             int col0, int k0) {
  int lane = threadIdx.x & 31;
  int hi = lane >> 4;
  const bf16_t* p = Bc + (size_t)(col0 + (lane & 15)) * ldb + k0 + hi * 16;
  v16bf b;
#pragma unroll
  for (int j = 0; j < 16; ++j) b[j] = p[j];
  return b;
}

// ---------- kernels ----------

__global__ void zero_f32_kernel(float* p, int n) {
  int i = blockIdx.x * blockDim.x + threadIdx.x;
  if (i < n) p[i] = 0.0f;
}

// conv weights [O][I][taps] fp32 -> [taps][O][I] bf16
__global__ void prep_convw_kernel(const float* __restrict__ w, bf16_t* __restrict__ wb,
                                  int taps) {
  int idx = blockIdx.x * blockDim.x + threadIdx.x;
  int total = H_ * H_ * taps;
  if (idx >= total) return;
  int k = idx % taps;
  int i = (idx / taps) % H_;
  int o = idx / (taps * H_);
  wb[((size_t)k * H_ + o) * H_ + i] = f2bf(w[idx]);
}

// linear weights [O][I] fp32 -> bf16 same layout
__global__ void prep_linw_kernel(const float* __restrict__ w, bf16_t* __restrict__ wb) {
  int idx = blockIdx.x * blockDim.x + threadIdx.x;
  if (idx >= H_ * H_) return;
  wb[idx] = f2bf(w[idx]);
}

// conv1: F0=40 in-channels, 5 taps, scalar fp32 (exact, only 1.3 GFLOP)
__global__ void conv1_kernel(const float* __restrict__ x, const float* __restrict__ w,
                             const float* __restrict__ bias, float* __restrict__ raw) {
  int idx = blockIdx.x * blockDim.x + threadIdx.x;
  if (idx >= B_ * T1_ * H_) return;
  int o = idx % H_;
  int t = (idx / H_) % T1_;
  int b = idx / (H_ * T1_);
  float acc = bias[o];
  const float* xp = x + ((size_t)b * T0_ + t) * F0_;
  const float* wp = w + (size_t)o * F0_ * 5;
#pragma unroll
  for (int k = 0; k < 5; ++k)
#pragma unroll
    for (int i = 0; i < F0_; ++i)
      acc += xp[k * F0_ + i] * wp[i * 5 + k];
  raw[idx] = acc;
}

// Unified dilated-conv / pointwise GEMM via bf16 WMMA with 4-way t-tile
// register blocking: one wave owns 16 o-channels x 64 timesteps, so each
// weight (A) fragment is reused by 4 B fragments.
// W [taps][H][H] bf16, X [B][TinP][H] bf16 (zero padded), raw out [B][Tv][H] fp32.
__global__ void conv_wmma_kernel(const bf16_t* __restrict__ W,
                                 const float* __restrict__ bias,
                                 const bf16_t* __restrict__ X,
                                 float* __restrict__ raw,
                                 int TinP, int Tv, int taps, int dil) {
  int o0 = blockIdx.x * 16;
  int t0 = blockIdx.y * 64;
  int b  = blockIdx.z;
  const bf16_t* Xb = X + (size_t)b * TinP * H_;
  v8f c0 = {}, c1 = {}, c2 = {}, c3 = {};
  for (int k = 0; k < taps; ++k) {
    int tshift = t0 + k * dil;
    const bf16_t* Wk = W + (size_t)k * H_ * H_;
#pragma unroll 2
    for (int i0 = 0; i0 < H_; i0 += 32) {
      v16bf a  = load_a_frag(Wk, H_, o0, i0);
      v16bf b0 = load_b_frag(Xb, H_, tshift +  0, i0);
      v16bf b1 = load_b_frag(Xb, H_, tshift + 16, i0);
      v16bf b2 = load_b_frag(Xb, H_, tshift + 32, i0);
      v16bf b3 = load_b_frag(Xb, H_, tshift + 48, i0);
      c0 = __builtin_amdgcn_wmma_f32_16x16x32_bf16(false, a, false, b0, (short)0, c0, false, false);
      c1 = __builtin_amdgcn_wmma_f32_16x16x32_bf16(false, a, false, b1, (short)0, c1, false, false);
      c2 = __builtin_amdgcn_wmma_f32_16x16x32_bf16(false, a, false, b2, (short)0, c2, false, false);
      c3 = __builtin_amdgcn_wmma_f32_16x16x32_bf16(false, a, false, b3, (short)0, c3, false, false);
    }
  }
  int lane = threadIdx.x & 31;
  int hi = lane >> 4;
  v8f cc[4] = {c0, c1, c2, c3};
#pragma unroll
  for (int g = 0; g < 4; ++g) {
    int t = t0 + g * 16 + (lane & 15);
    if (t < Tv) {
      float* outp = raw + ((size_t)b * Tv + t) * H_;
#pragma unroll
      for (int v = 0; v < 8; ++v) {
        int o = o0 + v + 8 * hi;
        outp[o] = cc[g][v] + bias[o];
      }
    }
  }
}

// Per-channel batch stats over raw [B][Tv][H]: mean, rsqrt(var+eps)
__global__ void bn_stats_kernel(const float* __restrict__ raw, int Tv,
                                float* __restrict__ mean, float* __restrict__ istd) {
  int o = blockIdx.x;
  int tid = threadIdx.x;
  int N = B_ * Tv;
  float s = 0.f, s2 = 0.f;
  for (int j = tid; j < N; j += blockDim.x) {
    float v = raw[(size_t)j * H_ + o];
    s += v; s2 += v * v;
  }
  __shared__ float sh[256], sh2[256];
  sh[tid] = s; sh2[tid] = s2;
  __syncthreads();
  for (int off = 128; off > 0; off >>= 1) {
    if (tid < off) { sh[tid] += sh[tid + off]; sh2[tid] += sh2[tid + off]; }
    __syncthreads();
  }
  if (tid == 0) {
    float m = sh[0] / (float)N;
    float var = sh2[0] / (float)N - m * m;
    mean[o] = m;
    istd[o] = rsqrtf(var + 1e-5f);
  }
}

// BN + relu20, write bf16 in [b][t][o] (zero fill t>=Tv up to TPout) and
// optionally transposed [b][o][t] (valid t only).
__global__ void bn_apply_kernel(const float* __restrict__ raw, int Tv, int TPout,
                                const float* __restrict__ mean,
                                const float* __restrict__ istd,
                                const float* __restrict__ g,
                                const float* __restrict__ be,
                                bf16_t* __restrict__ out_ti,
                                bf16_t* __restrict__ out_it) {
  int idx = blockIdx.x * blockDim.x + threadIdx.x;
  int total = B_ * TPout * H_;
  if (idx >= total) return;
  int o = idx % H_;
  int t = (idx / H_) % TPout;
  int b = idx / (H_ * TPout);
  float y = 0.f;
  if (t < Tv) {
    float v = raw[((size_t)b * Tv + t) * H_ + o];
    y = g[o] * (v - mean[o]) * istd[o] + be[o];
    y = fminf(fmaxf(y, 0.f), 20.f);
  }
  bf16_t yb = f2bf(y);
  if (out_ti) out_ti[idx] = yb;
  if (out_it && t < Tv) out_it[((size_t)b * H_ + o) * Tv + t] = yb;
}

// m[b,i,j] = sum_t x4[b,i,t]*x5[b,j,t]; both stored [b][c][t] bf16, ld=T3.
// One wave owns 16 i x 64 j (A fragment reused 4x).
__global__ void bmm_wmma_kernel(const bf16_t* __restrict__ Xc,
                                const bf16_t* __restrict__ Yc,
                                float* __restrict__ M) {
  int i0 = blockIdx.x * 16;
  int j0 = blockIdx.y * 64;
  int b  = blockIdx.z;
  const bf16_t* A  = Xc + (size_t)b * H_ * T3_;
  const bf16_t* Bc = Yc + (size_t)b * H_ * T3_;
  v8f c0 = {}, c1 = {}, c2 = {}, c3 = {};
#pragma unroll 2
  for (int k0 = 0; k0 < T3_; k0 += 32) {
    v16bf a  = load_a_frag(A, T3_, i0, k0);
    v16bf b0 = load_b_frag(Bc, T3_, j0 +  0, k0);
    v16bf b1 = load_b_frag(Bc, T3_, j0 + 16, k0);
    v16bf b2 = load_b_frag(Bc, T3_, j0 + 32, k0);
    v16bf b3 = load_b_frag(Bc, T3_, j0 + 48, k0);
    c0 = __builtin_amdgcn_wmma_f32_16x16x32_bf16(false, a, false, b0, (short)0, c0, false, false);
    c1 = __builtin_amdgcn_wmma_f32_16x16x32_bf16(false, a, false, b1, (short)0, c1, false, false);
    c2 = __builtin_amdgcn_wmma_f32_16x16x32_bf16(false, a, false, b2, (short)0, c2, false, false);
    c3 = __builtin_amdgcn_wmma_f32_16x16x32_bf16(false, a, false, b3, (short)0, c3, false, false);
  }
  int lane = threadIdx.x & 31;
  int hi = lane >> 4;
  float* outp = M + (size_t)b * KSQ_;
  v8f cc[4] = {c0, c1, c2, c3};
#pragma unroll
  for (int g = 0; g < 4; ++g) {
    int j = j0 + g * 16 + (lane & 15);
#pragma unroll
    for (int v = 0; v < 8; ++v)
      outp[(size_t)(i0 + v + 8 * hi) * H_ + j] = cc[g][v];
  }
}

// Bottleneck: hacc[16][512] += m[16][KSQ] x bot_w[512][KSQ]^T, fp32 WMMA.
// One wave owns the 16 batch rows x 64 n-columns; the m (A) fragment is
// loaded once per K step and reused by 4 bot_w fragments. bot_w (512MB)
// streams from HBM exactly once; K split across blocks, fp32 atomics merge.
#define BOT_KCHUNK 4096
#if __has_builtin(__builtin_amdgcn_wmma_f32_16x16x4_f32)
#define HAVE_WMMA_F32X4 1
__global__ void bot_wmma_kernel(const float* __restrict__ Mm,
                                const float* __restrict__ Wb,
                                float* __restrict__ hacc) {
  int n0 = blockIdx.x * 64;
  int kbase = blockIdx.y * BOT_KCHUNK;
  int lane = threadIdx.x & 31;
  int hi = lane >> 4;
  int r = lane & 15;
  const float* ap = Mm + (size_t)r * KSQ_ + 2 * hi;
  const float* bp0 = Wb + (size_t)(n0 +  0 + r) * KSQ_ + 2 * hi;
  const float* bp1 = Wb + (size_t)(n0 + 16 + r) * KSQ_ + 2 * hi;
  const float* bp2 = Wb + (size_t)(n0 + 32 + r) * KSQ_ + 2 * hi;
  const float* bp3 = Wb + (size_t)(n0 + 48 + r) * KSQ_ + 2 * hi;
  v8f c0 = {}, c1 = {}, c2 = {}, c3 = {};
#pragma unroll 4
  for (int k = kbase; k < kbase + BOT_KCHUNK; k += 4) {
    v2f a, b0, b1, b2, b3;
    a.x  = ap[k];  a.y  = ap[k + 1];
    b0.x = bp0[k]; b0.y = bp0[k + 1];
    b1.x = bp1[k]; b1.y = bp1[k + 1];
    b2.x = bp2[k]; b2.y = bp2[k + 1];
    b3.x = bp3[k]; b3.y = bp3[k + 1];
    c0 = __builtin_amdgcn_wmma_f32_16x16x4_f32(false, a, false, b0, (short)0, c0, false, false);
    c1 = __builtin_amdgcn_wmma_f32_16x16x4_f32(false, a, false, b1, (short)0, c1, false, false);
    c2 = __builtin_amdgcn_wmma_f32_16x16x4_f32(false, a, false, b2, (short)0, c2, false, false);
    c3 = __builtin_amdgcn_wmma_f32_16x16x4_f32(false, a, false, b3, (short)0, c3, false, false);
  }
  v8f cc[4] = {c0, c1, c2, c3};
#pragma unroll
  for (int g = 0; g < 4; ++g)
#pragma unroll
    for (int v = 0; v < 8; ++v)
      atomicAdd(&hacc[(v + 8 * hi) * H_ + n0 + g * 16 + r], cc[g][v]);
}
#else
__global__ void bot_valu_kernel(const float* __restrict__ Mm,
                                const float* __restrict__ Wb,
                                float* __restrict__ hacc) {
  int n = blockIdx.x;
  int kbase = blockIdx.y * BOT_KCHUNK;
  int tid = threadIdx.x;
  float acc[B_];
#pragma unroll
  for (int b = 0; b < B_; ++b) acc[b] = 0.f;
  for (int k = kbase + tid; k < kbase + BOT_KCHUNK; k += 256) {
    float w = Wb[(size_t)n * KSQ_ + k];
#pragma unroll
    for (int b = 0; b < B_; ++b) acc[b] += Mm[(size_t)b * KSQ_ + k] * w;
  }
  __shared__ float sh[256];
  for (int b = 0; b < B_; ++b) {
    sh[tid] = acc[b]; __syncthreads();
    for (int off = 128; off > 0; off >>= 1) {
      if (tid < off) sh[tid] += sh[tid + off];
      __syncthreads();
    }
    if (tid == 0) atomicAdd(&hacc[b * H_ + n], sh[0]);
    __syncthreads();
  }
}
#endif

// bn2d over batch (16) per feature + relu20
__global__ void bot_bn_kernel(const float* __restrict__ hacc,
                              const float* __restrict__ bias,
                              const float* __restrict__ g,
                              const float* __restrict__ be,
                              float* __restrict__ h) {
  int n = blockIdx.x * blockDim.x + threadIdx.x;
  if (n >= H_) return;
  float vals[B_]; float s = 0.f;
#pragma unroll
  for (int b = 0; b < B_; ++b) { float v = hacc[b * H_ + n] + bias[n]; vals[b] = v; s += v; }
  float m = s / (float)B_;
  float v2 = 0.f;
#pragma unroll
  for (int b = 0; b < B_; ++b) { float d = vals[b] - m; v2 += d * d; }
  float is = rsqrtf(v2 / (float)B_ + 1e-5f);
#pragma unroll
  for (int b = 0; b < B_; ++b) {
    float y = g[n] * (vals[b] - m) * is + be[n];
    h[b * H_ + n] = fminf(fmaxf(y, 0.f), 20.f);
  }
}

// emb = h @ emb_w^T + b, then L2-normalize * 10. One block per batch row.
__global__ void emb_norm_kernel(const float* __restrict__ h,
                                const float* __restrict__ Wemb,
                                const float* __restrict__ bemb,
                                float* __restrict__ out) {
  int b = blockIdx.x;
  int e = threadIdx.x;          // 512
  const float* hp = h + b * H_;
  const float* wp = Wemb + (size_t)e * H_;
  float acc = bemb[e];
  for (int i = 0; i < H_; ++i) acc += hp[i] * wp[i];
  __shared__ float sh[512];
  sh[e] = acc * acc;
  __syncthreads();
  for (int off = 256; off > 0; off >>= 1) {
    if (e < off) sh[e] += sh[e + off];
    __syncthreads();
  }
  float scale = 10.f * rsqrtf(sh[0] + 1e-10f);
  out[b * H_ + e] = acc * scale;
}

// ---------- host ----------

extern "C" void kernel_launch(void* const* d_in, const int* in_sizes, int n_in,
                              void* d_out, int out_size, void* d_ws, size_t ws_size,
                              hipStream_t stream) {
  (void)in_sizes; (void)n_in; (void)out_size; (void)ws_size;
  const float* input_x = (const float*)d_in[0];
  const float* conv1_w = (const float*)d_in[1];
  const float* conv1_b = (const float*)d_in[2];
  const float* bn1_g   = (const float*)d_in[3];
  const float* bn1_b   = (const float*)d_in[4];
  const float* conv2_w = (const float*)d_in[5];
  const float* conv2_b = (const float*)d_in[6];
  const float* bn2_g   = (const float*)d_in[7];
  const float* bn2_b   = (const float*)d_in[8];
  const float* conv3_w = (const float*)d_in[9];
  const float* conv3_b = (const float*)d_in[10];
  const float* bn3_g   = (const float*)d_in[11];
  const float* bn3_b   = (const float*)d_in[12];
  const float* lin4_w  = (const float*)d_in[13];
  const float* lin4_b  = (const float*)d_in[14];
  const float* bn4_g   = (const float*)d_in[15];
  const float* bn4_b   = (const float*)d_in[16];
  const float* lin5_w  = (const float*)d_in[17];
  const float* lin5_b  = (const float*)d_in[18];
  const float* bn5_g   = (const float*)d_in[19];
  const float* bn5_b   = (const float*)d_in[20];
  const float* bot_w   = (const float*)d_in[21];
  const float* bot_b   = (const float*)d_in[22];
  const float* bnb_g   = (const float*)d_in[23];
  const float* bnb_b   = (const float*)d_in[24];
  const float* emb_w   = (const float*)d_in[25];
  const float* emb_b   = (const float*)d_in[26];
  float* out = (float*)d_out;

  char* ws = (char*)d_ws;
  size_t off = 0;
  auto take = [&](size_t bytes) -> char* {
    char* p = ws + off;
    off += (bytes + 255) & ~(size_t)255;
    return p;
  };
  float* rawA = (float*)take((size_t)B_ * T1_ * H_ * 4);   // raw1 / raw3 / raw5
  float* rawB = (float*)take((size_t)B_ * T2_ * H_ * 4);   // raw2 / raw4
  float* mbuf = (float*)take((size_t)B_ * KSQ_ * 4);
  float* hacc = (float*)take((size_t)B_ * H_ * 4);
  float* hbuf = (float*)take((size_t)B_ * H_ * 4);
  float* mean = (float*)take(H_ * 4);
  float* istd = (float*)take(H_ * 4);
  bf16_t* bf1 = (bf16_t*)take((size_t)B_ * T1P_ * H_ * 2); // act1 / x4 [b][t][i]
  bf16_t* bf2 = (bf16_t*)take((size_t)B_ * T2P_ * H_ * 2); // act2 / x4 [b][i][t]
  bf16_t* bf3 = (bf16_t*)take((size_t)B_ * T3_  * H_ * 2); // act3 / x5 [b][j][t]
  bf16_t* w2b = (bf16_t*)take((size_t)3 * H_ * H_ * 2);
  bf16_t* w3b = (bf16_t*)take((size_t)3 * H_ * H_ * 2);
  bf16_t* l4b = (bf16_t*)take((size_t)H_ * H_ * 2);
  bf16_t* l5b = (bf16_t*)take((size_t)H_ * H_ * 2);

  // weight prep + zero accumulators
  prep_convw_kernel<<<(H_ * H_ * 3 + 255) / 256, 256, 0, stream>>>(conv2_w, w2b, 3);
  prep_convw_kernel<<<(H_ * H_ * 3 + 255) / 256, 256, 0, stream>>>(conv3_w, w3b, 3);
  prep_linw_kernel<<<(H_ * H_ + 255) / 256, 256, 0, stream>>>(lin4_w, l4b);
  prep_linw_kernel<<<(H_ * H_ + 255) / 256, 256, 0, stream>>>(lin5_w, l5b);
  zero_f32_kernel<<<(B_ * H_ + 255) / 256, 256, 0, stream>>>(hacc, B_ * H_);

  // layer 1: conv1 (scalar fp32) -> BN stats -> apply (zero-pad to T1P_)
  conv1_kernel<<<((size_t)B_ * T1_ * H_ + 255) / 256, 256, 0, stream>>>(
      input_x, conv1_w, conv1_b, rawA);
  bn_stats_kernel<<<H_, 256, 0, stream>>>(rawA, T1_, mean, istd);
  bn_apply_kernel<<<((size_t)B_ * T1P_ * H_ + 255) / 256, 256, 0, stream>>>(
      rawA, T1_, T1P_, mean, istd, bn1_g, bn1_b, bf1, nullptr);

  // layer 2: conv2 (3 taps, dil 2) WMMA; 7 groups of 64 t cover 392 (masked)
  conv_wmma_kernel<<<dim3(H_ / 16, 7, B_), 32, 0, stream>>>(
      w2b, conv2_b, bf1, rawB, T1P_, T2_, 3, 2);
  bn_stats_kernel<<<H_, 256, 0, stream>>>(rawB, T2_, mean, istd);
  bn_apply_kernel<<<((size_t)B_ * T2P_ * H_ + 255) / 256, 256, 0, stream>>>(
      rawB, T2_, T2P_, mean, istd, bn2_g, bn2_b, bf2, nullptr);

  // layer 3: conv3 (3 taps, dil 4) WMMA; 6 groups of 64 = 384 exact
  conv_wmma_kernel<<<dim3(H_ / 16, 6, B_), 32, 0, stream>>>(
      w3b, conv3_b, bf2, rawA, T2P_, T3_, 3, 4);
  bn_stats_kernel<<<H_, 256, 0, stream>>>(rawA, T3_, mean, istd);
  bn_apply_kernel<<<((size_t)B_ * T3_ * H_ + 255) / 256, 256, 0, stream>>>(
      rawA, T3_, T3_, mean, istd, bn3_g, bn3_b, bf3, nullptr);

  // layer 4: pointwise WMMA; apply writes [b][t][i] (bf1) and [b][i][t] (bf2)
  conv_wmma_kernel<<<dim3(H_ / 16, 6, B_), 32, 0, stream>>>(
      l4b, lin4_b, bf3, rawB, T3_, T3_, 1, 0);
  bn_stats_kernel<<<H_, 256, 0, stream>>>(rawB, T3_, mean, istd);
  bn_apply_kernel<<<((size_t)B_ * T3_ * H_ + 255) / 256, 256, 0, stream>>>(
      rawB, T3_, T3_, mean, istd, bn4_g, bn4_b, bf1, bf2);

  // layer 5: pointwise WMMA on x4; apply writes transposed [b][j][t] (bf3)
  conv_wmma_kernel<<<dim3(H_ / 16, 6, B_), 32, 0, stream>>>(
      l5b, lin5_b, bf1, rawA, T3_, T3_, 1, 0);
  bn_stats_kernel<<<H_, 256, 0, stream>>>(rawA, T3_, mean, istd);
  bn_apply_kernel<<<((size_t)B_ * T3_ * H_ + 255) / 256, 256, 0, stream>>>(
      rawA, T3_, T3_, mean, istd, bn5_g, bn5_b, nullptr, bf3);

  // batched outer-product GEMM: m[b] = x4[b] x x5[b]^T (8 j-groups of 64)
  bmm_wmma_kernel<<<dim3(H_ / 16, H_ / 64, B_), 32, 0, stream>>>(bf2, bf3, mbuf);

  // bottleneck: stream 512MB bot_w once; K-split partials via fp32 atomics
#if defined(HAVE_WMMA_F32X4)
  bot_wmma_kernel<<<dim3(H_ / 64, KSQ_ / BOT_KCHUNK), 32, 0, stream>>>(
      mbuf, bot_w, hacc);
#else
  bot_valu_kernel<<<dim3(H_, KSQ_ / BOT_KCHUNK), 256, 0, stream>>>(
      mbuf, bot_w, hacc);
#endif

  // bn2d + relu20, then emb GEMM + L2 normalize * 10
  bot_bn_kernel<<<(H_ + 255) / 256, 256, 0, stream>>>(hacc, bot_b, bnb_g, bnb_b, hbuf);
  emb_norm_kernel<<<B_, H_, 0, stream>>>(hbuf, emb_w, emb_b, out);
}
